// SAGEMLP_75307956568158
// MI455X (gfx1250) — matrix operand order repaired and checked by reading, not compiled, not run on hardware
//
#include <hip/hip_runtime.h>
#include <hip/hip_bf16.h>

typedef __attribute__((ext_vector_type(16))) _Float16 v16h;
typedef __attribute__((ext_vector_type(8)))  float    v8f;

__device__ __forceinline__ float gelu_f(float x) {
    return 0.5f * x * (1.0f + erff(x * 0.70710678118654752f));
}

// ---------------------------------------------------------------- utilities
__global__ void zero_f32(float* __restrict__ p, int n) {
    int i = blockIdx.x * 256 + threadIdx.x;
    if (i < n) p[i] = 0.0f;
}

__global__ void degree_kernel(const long long* __restrict__ ei, float* __restrict__ deg, int nE) {
    int e = blockIdx.x * 256 + threadIdx.x;
    if (e >= nE) return;
    long long d = ei[(long long)nE + e];
    atomicAdd(&deg[d], 1.0f);
}

__global__ void invdeg_kernel(const float* __restrict__ deg, float* __restrict__ invd, int nN) {
    int i = blockIdx.x * 256 + threadIdx.x;
    if (i < nN) invd[i] = 1.0f / fmaxf(deg[i], 1.0f);
}

// --------------------------------------------------- edge scatter (mean msg)
// 16 threads per edge, each moves 4 channels (float4 gather + 4 f32 atomics).
__global__ void scatter_add(const float* __restrict__ h, const long long* __restrict__ ei,
                            float* __restrict__ msg, int nE) {
    int gid = blockIdx.x * 256 + threadIdx.x;
    int e = gid >> 4;
    if (e >= nE) return;
    int c = (gid & 15) * 4;
    long long s = ei[e];
    long long d = ei[(long long)nE + e];
    float4 v = *(const float4*)(h + (size_t)s * 64 + c);
    float* p = msg + (size_t)d * 64 + c;
    atomicAdd(p + 0, v.x);
    atomicAdd(p + 1, v.y);
    atomicAdd(p + 2, v.z);
    atomicAdd(p + 3, v.w);
}

// -------------------------------------------- fused SAGE layer (WMMA GEMM)
// block = 256 threads (8 waves). Each wave: 16 nodes x 64 cols.
// f = (msg*invd) @ Wl + bl + h @ Wr ; gelu ; LayerNorm(C=64) ; +h residual.
__global__ __launch_bounds__(256) void sage_layer(
    const float* __restrict__ hIn, const float* __restrict__ msg, const float* __restrict__ invDeg,
    const float* __restrict__ Wl, const float* __restrict__ bl, const float* __restrict__ Wr,
    const float* __restrict__ lng, const float* __restrict__ lnb,
    float* __restrict__ hOut, int nN)
{
    __shared__ __align__(32) _Float16 sWl[4096];
    __shared__ __align__(32) _Float16 sWr[4096];
    __shared__ float fT[128 * 68];

    const int tid = threadIdx.x;

    // Stage weights into LDS pre-swizzled in B-fragment lane order:
    // fragment(kc,n): lane L holds col = n*16+(L&15), k = kc*32 + j + (L>=16?16:0)
    for (int idx = tid; idx < 4096; idx += 256) {
        int j    = idx & 15;
        int ln   = (idx >> 4) & 31;
        int frag = idx >> 9;                 // 0..7 = kc*4+n
        int kc = frag >> 2, n = frag & 3;
        int k   = kc * 32 + j + ((ln >= 16) ? 16 : 0);
        int col = n * 16 + (ln & 15);
        sWl[idx] = (_Float16)Wl[k * 64 + col];
        sWr[idx] = (_Float16)Wr[k * 64 + col];
    }
    __syncthreads();

    const int wave = tid >> 5, lane = tid & 31;
    const int r0 = blockIdx.x * 128 + wave * 16;
    const int row = r0 + (lane & 15);
    const bool rv = row < nN;
    const float inv = rv ? invDeg[row] : 0.0f;
    const float* mrow = msg + (size_t)row * 64;
    const float* hrow = hIn + (size_t)row * 64;

    v8f acc[4] = {};

    #pragma unroll
    for (int kc = 0; kc < 2; kc++) {
        const int kb = kc * 32 + ((lane >= 16) ? 16 : 0);
        v16h aAgg = {}, aH = {};
        if (rv) {
            float4 a0 = *(const float4*)(mrow + kb);
            float4 a1 = *(const float4*)(mrow + kb + 4);
            float4 a2 = *(const float4*)(mrow + kb + 8);
            float4 a3 = *(const float4*)(mrow + kb + 12);
            float4 b0 = *(const float4*)(hrow + kb);
            float4 b1 = *(const float4*)(hrow + kb + 4);
            float4 b2 = *(const float4*)(hrow + kb + 8);
            float4 b3 = *(const float4*)(hrow + kb + 12);
            aAgg[0]  = (_Float16)(a0.x * inv); aAgg[1]  = (_Float16)(a0.y * inv);
            aAgg[2]  = (_Float16)(a0.z * inv); aAgg[3]  = (_Float16)(a0.w * inv);
            aAgg[4]  = (_Float16)(a1.x * inv); aAgg[5]  = (_Float16)(a1.y * inv);
            aAgg[6]  = (_Float16)(a1.z * inv); aAgg[7]  = (_Float16)(a1.w * inv);
            aAgg[8]  = (_Float16)(a2.x * inv); aAgg[9]  = (_Float16)(a2.y * inv);
            aAgg[10] = (_Float16)(a2.z * inv); aAgg[11] = (_Float16)(a2.w * inv);
            aAgg[12] = (_Float16)(a3.x * inv); aAgg[13] = (_Float16)(a3.y * inv);
            aAgg[14] = (_Float16)(a3.z * inv); aAgg[15] = (_Float16)(a3.w * inv);
            aH[0]  = (_Float16)b0.x; aH[1]  = (_Float16)b0.y;
            aH[2]  = (_Float16)b0.z; aH[3]  = (_Float16)b0.w;
            aH[4]  = (_Float16)b1.x; aH[5]  = (_Float16)b1.y;
            aH[6]  = (_Float16)b1.z; aH[7]  = (_Float16)b1.w;
            aH[8]  = (_Float16)b2.x; aH[9]  = (_Float16)b2.y;
            aH[10] = (_Float16)b2.z; aH[11] = (_Float16)b2.w;
            aH[12] = (_Float16)b3.x; aH[13] = (_Float16)b3.y;
            aH[14] = (_Float16)b3.z; aH[15] = (_Float16)b3.w;
        }
        #pragma unroll
        for (int n = 0; n < 4; n++) {
            v16h bL = *(const v16h*)&sWl[((kc * 4 + n) * 32 + lane) * 16];
            v16h bR = *(const v16h*)&sWr[((kc * 4 + n) * 32 + lane) * 16];
            acc[n] = __builtin_amdgcn_wmma_f32_16x16x32_f16(
                false, aAgg, false, bL, (short)0, acc[n], false, false);
            acc[n] = __builtin_amdgcn_wmma_f32_16x16x32_f16(
                false, aH,   false, bR, (short)0, acc[n], false, false);
        }
    }

    // bias + gelu -> LDS tile (C/D layout: lane L holds col=n*16+(L&15), M=v+(L>=16?8:0))
    const int colBase = lane & 15;
    const int mAdd = (lane >= 16) ? 8 : 0;
    #pragma unroll
    for (int n = 0; n < 4; n++) {
        int col = n * 16 + colBase;
        float bias = bl[col];
        #pragma unroll
        for (int v = 0; v < 8; v++) {
            float f = acc[n][v] + bias;
            fT[(wave * 16 + v + mAdd) * 68 + col] = gelu_f(f);
        }
    }
    __syncthreads();

    // LayerNorm over 64 channels + residual; 2 threads per row
    const int rl = tid >> 1;
    const int half = tid & 1;
    const int grow = blockIdx.x * 128 + rl;
    if (grow < nN) {
        const float* fr = &fT[rl * 68 + half * 32];
        float s = 0.0f, s2 = 0.0f;
        #pragma unroll
        for (int c = 0; c < 32; c++) { float v = fr[c]; s += v; s2 += v * v; }
        s  += __shfl_xor(s, 1, 32);
        s2 += __shfl_xor(s2, 1, 32);
        float mu  = s * (1.0f / 64.0f);
        float var = s2 * (1.0f / 64.0f) - mu * mu;
        float rstd = rsqrtf(var + 1e-5f);
        const float* hr = hIn + (size_t)grow * 64 + half * 32;
        float* ho = hOut + (size_t)grow * 64 + half * 32;
        const float* gg = lng + half * 32;
        const float* bb = lnb + half * 32;
        #pragma unroll
        for (int c = 0; c < 32; c++)
            ho[c] = (fr[c] - mu) * rstd * gg[c] + bb[c] + hr[c];
    }
}

// ------------------------------------------------------- segment-max pooling
__device__ __forceinline__ unsigned fmap(float f) {
    unsigned b = __float_as_uint(f);
    return (b & 0x80000000u) ? ~b : (b | 0x80000000u);
}

__global__ void pool_max(const float* __restrict__ h, const long long* __restrict__ batch,
                         unsigned* __restrict__ pooledU, int nN) {
    __shared__ unsigned loc[1024];  // 16 graphs x 64 ch
    const int t = threadIdx.x;
    for (int i = t; i < 1024; i += 256) loc[i] = 0u;
    __syncthreads();
    const int start = blockIdx.x * 1024;
    const int end = min(start + 1024, nN);
    const int ch = t & 63;
    for (int node = start + (t >> 6); node < end; node += 4) {
        int g = (int)batch[node];
        float v = h[(size_t)node * 64 + ch];
        atomicMax(&loc[g * 64 + ch], fmap(v));
    }
    __syncthreads();
    for (int i = t; i < 1024; i += 256)
        if (loc[i]) atomicMax(&pooledU[i], loc[i]);
}

// ----------------------------------------------------------------- MLP head
__global__ __launch_bounds__(512) void mlp_head(
    const unsigned* __restrict__ pooledU,
    const float* __restrict__ w0, const float* __restrict__ b0,
    const float* __restrict__ g0, const float* __restrict__ bn0,
    const float* __restrict__ w,  const float* __restrict__ b,
    const float* __restrict__ g,  const float* __restrict__ bn,
    const float* __restrict__ hw, const float* __restrict__ hb,
    float* __restrict__ out)
{
    __shared__ float p[16][64];
    __shared__ float zb[16][32];
    const int j = threadIdx.x;     // channel 0..31 (one wave per graph)
    const int gi = threadIdx.y;    // graph 0..15
    const int tid = gi * 32 + j;

    for (int i = tid; i < 1024; i += 512) {
        unsigned m = pooledU[i];
        unsigned bits = (m & 0x80000000u) ? (m & 0x7FFFFFFFu) : ~m;
        p[i >> 6][i & 63] = __uint_as_float(bits);
    }
    __syncthreads();

    float acc = b0[j];
    #pragma unroll
    for (int k = 0; k < 64; k++) acc += p[gi][k] * w0[k * 32 + j];
    acc = gelu_f(acc);
    float s = acc, s2 = acc * acc;
    #pragma unroll
    for (int m = 16; m >= 1; m >>= 1) { s += __shfl_xor(s, m, 32); s2 += __shfl_xor(s2, m, 32); }
    float mu = s * (1.0f / 32.0f), var = s2 * (1.0f / 32.0f) - mu * mu;
    float z = (acc - mu) * rsqrtf(var + 1e-5f) * g0[j] + bn0[j];

    for (int i = 0; i < 3; i++) {
        zb[gi][j] = z;
        __syncthreads();
        float t = b[i * 32 + j];
        #pragma unroll
        for (int k = 0; k < 32; k++) t += zb[gi][k] * w[i * 1024 + k * 32 + j];
        t = gelu_f(t);
        float ts = t, ts2 = t * t;
        #pragma unroll
        for (int m = 16; m >= 1; m >>= 1) { ts += __shfl_xor(ts, m, 32); ts2 += __shfl_xor(ts2, m, 32); }
        float mu2 = ts * (1.0f / 32.0f), var2 = ts2 * (1.0f / 32.0f) - mu2 * mu2;
        z = (t - mu2) * rsqrtf(var2 + 1e-5f) * g[i * 32 + j] + bn[i * 32 + j] + z;
        __syncthreads();
    }

    float hv = z * hw[j];
    #pragma unroll
    for (int m = 16; m >= 1; m >>= 1) hv += __shfl_xor(hv, m, 32);
    if (j == 0) out[gi] = hv + hb[0];
}

// ------------------------------------------------------------------- launch
extern "C" void kernel_launch(void* const* d_in, const int* in_sizes, int n_in,
                              void* d_out, int out_size, void* d_ws, size_t ws_size,
                              hipStream_t stream) {
    const float*     x     = (const float*)d_in[0];
    const long long* ei    = (const long long*)d_in[1];
    const long long* batch = (const long long*)d_in[2];
    const float* sWl = (const float*)d_in[3];
    const float* sbl = (const float*)d_in[4];
    const float* sWr = (const float*)d_in[5];
    const float* sg  = (const float*)d_in[6];
    const float* sb  = (const float*)d_in[7];
    const float* m0W = (const float*)d_in[8];
    const float* m0b = (const float*)d_in[9];
    const float* m0g = (const float*)d_in[10];
    const float* m0n = (const float*)d_in[11];
    const float* mW  = (const float*)d_in[12];
    const float* mb  = (const float*)d_in[13];
    const float* mg  = (const float*)d_in[14];
    const float* mn  = (const float*)d_in[15];
    const float* hW  = (const float*)d_in[16];
    const float* hb  = (const float*)d_in[17];
    float* out = (float*)d_out;

    const int nN = in_sizes[0] / 64;   // 100000
    const int nE = in_sizes[1] / 2;    // 3200000
    const size_t NC = (size_t)nN * 64;

    float* ws   = (float*)d_ws;
    float* msg  = ws;
    float* hA   = ws + NC;
    float* hB   = ws + 2 * NC;
    float* deg  = ws + 3 * NC;
    float* invd = deg + nN;
    unsigned* pooledU = (unsigned*)(invd + nN);

    // degrees -> inv_deg (once)
    zero_f32<<<(nN + 255) / 256, 256, 0, stream>>>(deg, nN);
    degree_kernel<<<(nE + 255) / 256, 256, 0, stream>>>(ei, deg, nE);
    invdeg_kernel<<<(nN + 255) / 256, 256, 0, stream>>>(deg, invd, nN);

    const float* hc = x;
    float* hn = hA;
    for (int l = 0; l < 6; l++) {
        zero_f32<<<((int)NC + 255) / 256, 256, 0, stream>>>(msg, (int)NC);
        scatter_add<<<(nE * 16 + 255) / 256, 256, 0, stream>>>(hc, ei, msg, nE);
        sage_layer<<<(nN + 127) / 128, 256, 0, stream>>>(
            hc, msg, invd,
            sWl + (size_t)l * 4096, sbl + (size_t)l * 64, sWr + (size_t)l * 4096,
            sg + (size_t)l * 64, sb + (size_t)l * 64, hn, nN);
        hc = hn;
        hn = (hn == hA) ? hB : hA;
    }

    zero_f32<<<4, 256, 0, stream>>>((float*)pooledU, 1024);
    pool_max<<<(nN + 1023) / 1024, 256, 0, stream>>>(hc, batch, pooledU, nN);
    mlp_head<<<1, dim3(32, 16), 0, stream>>>(pooledU, m0W, m0b, m0g, m0n,
                                             mW, mb, mg, mn, hW, hb, out);
}
